// SENNGC_25366076850341
// MI455X (gfx1250) — matrix operand order repaired and checked by reading, not compiled, not run on hardware
//
#include <hip/hip_runtime.h>
#include <hip/hip_bf16.h>
#include <math.h>

#ifndef __has_builtin
#define __has_builtin(x) 0
#endif
#if __has_builtin(__builtin_amdgcn_tensor_load_to_lds)
#define HAVE_TDM 1
#else
#define HAVE_TDM 0
#endif
#if __has_builtin(__builtin_amdgcn_s_wait_tensorcnt)
#define HAVE_TWAIT 1
#else
#define HAVE_TWAIT 0
#endif
#if defined(__has_include)
#if __has_include(<hip/amd_detail/amd_gfx1250_TDM.h>)
#define TDM_ARITY6 1
#else
#define TDM_ARITY6 0
#endif
#else
#define TDM_ARITY6 0
#endif

// ---------------- problem constants ----------------
#define T_TOTAL   20000
#define KNUM      5
#define PNUM      10
#define HDIM      256
#define NDX       20          // ND
#define NDPAD     32          // ND padded to WMMA K=32
#define OUTX      40          // OUT
#define OUTPAD    48          // OUT padded to 3 N-tiles
#define IND       40          // INPUT_DIM
#define ROWSTR    (KNUM * PNUM * IND)   // 2000 floats per t-row of inputs
#define MTILE     128

// ---------------- d_out layout (float elements) ----------------
static constexpr size_t PRED_OFF = 0;                                            // T*20
static constexpr size_t COEF_OFF = (size_t)T_TOTAL * PNUM * 2;                   // 400,000
static constexpr size_t FP_OFF   = COEF_OFF + (size_t)T_TOTAL * KNUM * PNUM * 2 * NDX; // 40,400,000

// ---------------- workspace layout (bytes) ----------------
static constexpr size_t W1T_OFF   = 0;          // bf16 [kp][h=256][kd=32]
static constexpr size_t W2T_OFF   = 819200;     // bf16 [kp][g=256][h=256]
static constexpr size_t W3T_OFF   = 7372800;    // bf16 [kp][o=48][h=256]
static constexpr size_t PREDP_OFF = 8601600;    // f32  [t][k][pd=20]

// ---------------- LDS layout (bytes, dynamic) ----------------
static constexpr unsigned LDS_X   = 0;                         // 128*32 bf16 =   8 KB
static constexpr unsigned LDS_H1  = 8192;                      // 128*256 bf16 = 64 KB
static constexpr unsigned LDS_H2  = 8192 + 65536;              // 128*256 bf16 = 64 KB
static constexpr unsigned LDS_W2  = 8192 + 2 * 65536;          // 256*256 bf16 = 128 KB
static constexpr unsigned LDS_W13 = LDS_W2 + 131072;           // W1 16 KB + W3 24 KB
static constexpr unsigned LDS_BYTES = LDS_W13 + 40960;         // 311,296 B <= 320 KB/WGP

typedef __attribute__((ext_vector_type(16))) __bf16 v16bf;
typedef __attribute__((ext_vector_type(8)))  float  v8f;
typedef unsigned int u32x4 __attribute__((ext_vector_type(4)));
typedef int          i32x4 __attribute__((ext_vector_type(4)));
typedef int          i32x8 __attribute__((ext_vector_type(8)));

union FragU { v16bf v; uint4 q[2]; };

// A-matrix 16x32 bf16 fragment from LDS, per CDNA5 layout:
// lanes 0-15: row M=lane, K = {0..7, 16..23}; lanes 16-31: row M=lane-16, K = {8..15, 24..31}
// Per lane this is two contiguous 16-byte chunks -> 2x ds_load_b128.
__device__ __forceinline__ v16bf frag_a(const __bf16* base, int strideElems, int row, int koff, int half) {
  FragU f;
  const char* bp = (const char*)(base + (size_t)row * strideElems + koff);
  f.q[0] = *(const uint4*)(bp + half * 16);
  f.q[1] = *(const uint4*)(bp + 32 + half * 16);
  return f.v;
}

// B-matrix 32x16 bf16 fragment from LDS stored [N][K]:
// lanes 0-15: col N=lane, K = {0..7, 16..23}; lanes 16-31: col N=lane-16, K = {8..15, 24..31}
// Per lane: two contiguous 16-byte chunks -> 2x ds_load_b128.
__device__ __forceinline__ v16bf frag_b(const __bf16* base, int strideElems, int col, int koff, int half) {
  FragU f;
  const char* bp = (const char*)(base + (size_t)col * strideElems + koff);
  f.q[0] = *(const uint4*)(bp + half * 16);
  f.q[1] = *(const uint4*)(bp + 32 + half * 16);
  return f.v;
}

#if HAVE_TDM
// 1-D contiguous TDM copy: global -> LDS, nbytes multiple of 8, nbytes/8 <= 65535.
__device__ __forceinline__ void tdm_load_1d(unsigned ldsOff, const void* gsrc, unsigned nbytes) {
  unsigned long long ga = (unsigned long long)gsrc;
  unsigned n8 = nbytes >> 3;                    // elements of 8 bytes
  u32x4 g0;
  g0[0] = 1u;                                    // count=1 (valid), user mode
  g0[1] = ldsOff;                                // lds_addr (bytes)
  g0[2] = (unsigned)(ga & 0xffffffffu);          // global_addr[31:0]
  g0[3] = (unsigned)((ga >> 32) & 0x1ffffffu) | (2u << 30); // global_addr[56:32], type=2
  i32x8 g1;
  g1[0] = (int)(3u << 16);                       // workgroup_mask=0, data_size=3 (8B)
  g1[1] = (int)((n8 & 0xffffu) << 16);           // tensor_dim0[15:0] at bits[63:48]
  g1[2] = (int)(((n8 >> 16) & 0xffffu) | (1u << 16)); // tensor_dim0[31:16], tensor_dim1=1
  g1[3] = (int)((n8 & 0xffffu) << 16);           // tile_dim0 at bits[127:112]
  g1[4] = 1;                                     // tile_dim1=1, tile_dim2=0
  g1[5] = (int)n8;                               // tensor_dim0_stride[31:0]
  g1[6] = 0;
  g1[7] = 0;
  i32x4 z4 = {0, 0, 0, 0};
#if TDM_ARITY6
  i32x8 z8 = {0, 0, 0, 0, 0, 0, 0, 0};
  __builtin_amdgcn_tensor_load_to_lds(g0, g1, z4, z4, z8, 0);
#else
  __builtin_amdgcn_tensor_load_to_lds(g0, g1, z4, z4, 0);
#endif
}
#endif

// ---------------- prep: f32 weights -> bf16, transposed to [N][K] ----------------
__global__ void conv_w1(const float* __restrict__ W1, __bf16* __restrict__ w1t) {
  int i = blockIdx.x * blockDim.x + threadIdx.x;
  if (i >= KNUM * PNUM * HDIM * NDPAD) return;
  int kd = i % NDPAD; int rest = i / NDPAD; int h = rest % HDIM; int kp = rest / HDIM;
  float v = 0.f;
  if (kd < NDX) v = W1[((size_t)kp * NDX + kd) * HDIM + h];
  w1t[i] = (__bf16)v;
}
__global__ void conv_w2(const float* __restrict__ W2, __bf16* __restrict__ w2t) {
  int i = blockIdx.x * blockDim.x + threadIdx.x;
  if (i >= KNUM * PNUM * HDIM * HDIM) return;
  int hk = i % HDIM; int rest = i / HDIM; int g = rest % HDIM; int kp = rest / HDIM;
  w2t[i] = (__bf16)W2[((size_t)kp * HDIM + hk) * HDIM + g];
}
__global__ void conv_w3(const float* __restrict__ W3, __bf16* __restrict__ w3t) {
  int i = blockIdx.x * blockDim.x + threadIdx.x;
  if (i >= KNUM * PNUM * OUTPAD * HDIM) return;
  int hk = i % HDIM; int rest = i / HDIM; int o = rest % OUTPAD; int kp = rest / OUTPAD;
  float v = 0.f;
  if (o < OUTX) v = W3[((size_t)kp * HDIM + hk) * OUTX + o];
  w3t[i] = (__bf16)v;
}

// ---------------- coeffs_FP (sharp sigmoids, f32 VALU) ----------------
__global__ void gate_kernel(const float* __restrict__ inp, const float* __restrict__ avoidp,
                            float* __restrict__ outFP) {
  int i = blockIdx.x * blockDim.x + threadIdx.x;
  if (i >= T_TOTAL * PNUM * 9) return;
  int t = i / (PNUM * 9); int r = i % (PNUM * 9); int p = r / 9; int j = r % 9;
  const float* row = inp + (size_t)t * ROWSTR + (KNUM - 1) * PNUM * IND; // inputs[:, -1, :]
  float xF  = row[p * IND + 31 + j];
  float xF2 = row[p * IND + 22 + j];
  float a = *avoidp;
  float s1 = 1.f / (1.f + __expf(-100.f * (xF - a)));
  float s2 = 1.f / (1.f + __expf(-100.f * xF2));
  outFP[i] = s1 * (s2 - 0.5f) * 2.f;
}

// ---------------- fused MLP: one workgroup = 128 t-rows x one (k,p) ----------------
__global__ __launch_bounds__(256)
void mlp_fused(const float* __restrict__ inp,
               const __bf16* __restrict__ w1t, const __bf16* __restrict__ w2t,
               const __bf16* __restrict__ w3t,
               const float* __restrict__ b1, const float* __restrict__ b2,
               const float* __restrict__ b3,
               float* __restrict__ out, float* __restrict__ predp) {
  extern __shared__ char smem[];
  __bf16* lds_x   = (__bf16*)(smem + LDS_X);
  __bf16* lds_h1  = (__bf16*)(smem + LDS_H1);
  __bf16* lds_h2  = (__bf16*)(smem + LDS_H2);
  __bf16* lds_w2  = (__bf16*)(smem + LDS_W2);
  __bf16* lds_w1  = (__bf16*)(smem + LDS_W13);           // 16 KB
  __bf16* lds_w3  = (__bf16*)(smem + LDS_W13 + 16384);   // 24 KB
  float*  lds_c2  = (float*)(smem + LDS_H1);             // reuse h1 region after L2

  const int tid  = threadIdx.x;
  const int lane = tid & 31;
  const int wave = tid >> 5;
  const int kp   = blockIdx.y;
  const int k    = kp / PNUM;
  const int p    = kp % PNUM;
  const int m0   = blockIdx.x * MTILE;
  const int nlo  = lane & 15;
  const int half = lane >> 4;
  const int rbase = half * 8;

  // ---- stage all weight panels: TDM DMA (one issuing wave) or vector-copy fallback ----
#if HAVE_TDM
  if (tid == 0) {
    tdm_load_1d(LDS_W13,         w1t + (size_t)kp * HDIM * NDPAD,  HDIM * NDPAD * 2);  // 16 KB
    tdm_load_1d(LDS_W2,          w2t + (size_t)kp * HDIM * HDIM,   HDIM * HDIM * 2);   // 128 KB
    tdm_load_1d(LDS_W13 + 16384, w3t + (size_t)kp * OUTPAD * HDIM, OUTPAD * HDIM * 2); // 24 KB
  }
#else
  {
    const uint4* s1p = (const uint4*)(w1t + (size_t)kp * HDIM * NDPAD);
    const uint4* s2p = (const uint4*)(w2t + (size_t)kp * HDIM * HDIM);
    const uint4* s3p = (const uint4*)(w3t + (size_t)kp * OUTPAD * HDIM);
    uint4* d1p = (uint4*)lds_w1; uint4* d2p = (uint4*)lds_w2; uint4* d3p = (uint4*)lds_w3;
    #pragma unroll
    for (int i = 0; i < 4; ++i)  d1p[tid + i * 256] = s1p[tid + i * 256];
    #pragma unroll
    for (int i = 0; i < 32; ++i) d2p[tid + i * 256] = s2p[tid + i * 256];
    #pragma unroll
    for (int i = 0; i < 6; ++i)  d3p[tid + i * 256] = s3p[tid + i * 256];
  }
#endif

  // ---- stage X tile (overlaps TDM): gather sel (20 cols), bf16, pad K to 32 ----
  {
    int r = tid >> 1;
    int c0 = (tid & 1) * 16;
    int t = m0 + r;
    const float* rowp = inp + (size_t)t * ROWSTR + k * PNUM * IND + p * IND;
    #pragma unroll
    for (int c = 0; c < 16; ++c) {
      int col = c0 + c;
      float v = 0.f;
      if (col < NDX && t < T_TOTAL) {
        int cc = (col < 2) ? col : (col + 2);   // gather idx: p*40 + {0,1,4..21}
        v = rowp[cc];
      }
      lds_x[r * NDPAD + col] = (__bf16)v;
    }
  }
#if HAVE_TDM && HAVE_TWAIT
  __builtin_amdgcn_s_wait_tensorcnt(2);   // W1 (first issued) complete
#endif
  __syncthreads();

  // ---- L1: h1 = relu(X @ W1 + b1) ----
  {
    v16bf a = frag_a(lds_x, NDPAD, wave * 16 + nlo, 0, half);
    const float* b1p = b1 + (size_t)kp * HDIM;
    #pragma unroll
    for (int nt = 0; nt < 16; ++nt) {
      v8f acc = {0.f, 0.f, 0.f, 0.f, 0.f, 0.f, 0.f, 0.f};
      v16bf bf = frag_b(lds_w1, NDPAD, nt * 16 + nlo, 0, half);
      acc = __builtin_amdgcn_wmma_f32_16x16x32_bf16(false, a, false, bf, (short)0, acc, false, false);
      int col = nt * 16 + nlo;
      float bias = b1p[col];
      #pragma unroll
      for (int v = 0; v < 8; ++v) {
        float hv = fmaxf(acc[v] + bias, 0.f);
        lds_h1[(wave * 16 + rbase + v) * HDIM + col] = (__bf16)hv;
      }
    }
  }
#if HAVE_TDM && HAVE_TWAIT
  __builtin_amdgcn_s_wait_tensorcnt(1);   // W2 complete
#endif
  __syncthreads();

  // ---- L2: h2 = relu(h1 @ W2 + b2): 8 K-steps x 16 N-tiles = 128 WMMAs/wave ----
  {
    v8f acc[16];
    #pragma unroll
    for (int nt = 0; nt < 16; ++nt) acc[nt] = (v8f){0.f, 0.f, 0.f, 0.f, 0.f, 0.f, 0.f, 0.f};
    for (int kk = 0; kk < HDIM; kk += 32) {
      v16bf a = frag_a(lds_h1, HDIM, wave * 16 + nlo, kk, half);
      #pragma unroll
      for (int nt = 0; nt < 16; ++nt) {
        v16bf bf = frag_b(lds_w2, HDIM, nt * 16 + nlo, kk, half);
        acc[nt] = __builtin_amdgcn_wmma_f32_16x16x32_bf16(false, a, false, bf, (short)0, acc[nt], false, false);
      }
    }
    const float* b2p = b2 + (size_t)kp * HDIM;
    #pragma unroll
    for (int nt = 0; nt < 16; ++nt) {
      int col = nt * 16 + nlo;
      float bias = b2p[col];
      #pragma unroll
      for (int v = 0; v < 8; ++v) {
        float hv = fmaxf(acc[nt][v] + bias, 0.f);
        lds_h2[(wave * 16 + rbase + v) * HDIM + col] = (__bf16)hv;
      }
    }
  }
#if HAVE_TDM && HAVE_TWAIT
  __builtin_amdgcn_s_wait_tensorcnt(0);   // W3 complete
#endif
  __syncthreads();

  // ---- L3: c = h2 @ W3 + b3; square + gate -> lds_c2 ----
  {
    v8f acc[3];
    #pragma unroll
    for (int nt = 0; nt < 3; ++nt) acc[nt] = (v8f){0.f, 0.f, 0.f, 0.f, 0.f, 0.f, 0.f, 0.f};
    for (int kk = 0; kk < HDIM; kk += 32) {
      v16bf a = frag_a(lds_h2, HDIM, wave * 16 + nlo, kk, half);
      #pragma unroll
      for (int nt = 0; nt < 3; ++nt) {
        v16bf bf = frag_b(lds_w3, HDIM, nt * 16 + nlo, kk, half);
        acc[nt] = __builtin_amdgcn_wmma_f32_16x16x32_bf16(false, a, false, bf, (short)0, acc[nt], false, false);
      }
    }
    const float* b3p = b3 + (size_t)kp * OUTX;
    const float* fp  = out + FP_OFF;
    #pragma unroll
    for (int nt = 0; nt < 3; ++nt) {
      int o = nt * 16 + nlo;            // 0..47, valid < 40; o = d*20 + n
      bool valid = (o < OUTX);
      float bias = valid ? b3p[o] : 0.f;
      int n = valid ? (o % NDX) : 0;
      #pragma unroll
      for (int v = 0; v < 8; ++v) {
        int rl = wave * 16 + rbase + v;
        int t = m0 + rl; if (t >= T_TOTAL) t = T_TOTAL - 1;
        float c = acc[nt][v] + bias;
        float m = 1.f;
        if (valid && n >= 2) m = fp[((size_t)t * PNUM + p) * 9 + ((n - 2) >> 1)];
        if (valid) lds_c2[rl * OUTPAD + o] = c * c * m;
      }
    }
  }
  __syncthreads();

  // ---- coeffs store (f32, 20 contiguous) + preds partial (sum over n) ----
  {
    int rl = tid >> 1;
    int d  = tid & 1;
    int t  = m0 + rl;
    if (t < T_TOTAL) {
      const float* src = lds_c2 + rl * OUTPAD + d * NDX;
      float* dst = out + COEF_OFF + (((size_t)t * KNUM + k) * (PNUM * 2) + p * 2 + d) * NDX;
      const float* rowp = inp + (size_t)t * ROWSTR + k * PNUM * IND + p * IND;
      float s = 0.f;
      #pragma unroll
      for (int q = 0; q < 5; ++q) {
        float4 vv = *(const float4*)(src + q * 4);
        *(float4*)(dst + q * 4) = vv;
        #pragma unroll
        for (int e = 0; e < 4; ++e) {
          int n = q * 4 + e;
          int cc = (n < 2) ? n : (n + 2);
          s += (&vv.x)[e] * rowp[cc];
        }
      }
      predp[((size_t)t * KNUM + k) * (PNUM * 2) + p * 2 + d] = s;
    }
  }
}

// ---------------- preds: deterministic reduction over k ----------------
__global__ void pred_reduce(const float* __restrict__ partial, float* __restrict__ out) {
  int i = blockIdx.x * blockDim.x + threadIdx.x;
  if (i >= T_TOTAL * PNUM * 2) return;
  int t = i / (PNUM * 2), pd = i % (PNUM * 2);
  float s = 0.f;
  #pragma unroll
  for (int k = 0; k < KNUM; ++k) s += partial[((size_t)t * KNUM + k) * (PNUM * 2) + pd];
  out[PRED_OFF + i] = s;
}

extern "C" void kernel_launch(void* const* d_in, const int* in_sizes, int n_in,
                              void* d_out, int out_size, void* d_ws, size_t ws_size,
                              hipStream_t stream) {
  const float* inp   = (const float*)d_in[0];
  const float* W1    = (const float*)d_in[1];
  const float* b1    = (const float*)d_in[2];
  const float* W2    = (const float*)d_in[3];
  const float* b2    = (const float*)d_in[4];
  const float* W3    = (const float*)d_in[5];
  const float* b3    = (const float*)d_in[6];
  const float* avoid = (const float*)d_in[7];
  float* out = (float*)d_out;
  char*  ws  = (char*)d_ws;

  __bf16* w1t  = (__bf16*)(ws + W1T_OFF);
  __bf16* w2t  = (__bf16*)(ws + W2T_OFF);
  __bf16* w3t  = (__bf16*)(ws + W3T_OFF);
  float* predp = (float*)(ws + PREDP_OFF);

  {
    int n = KNUM * PNUM * HDIM * NDPAD;
    conv_w1<<<(n + 255) / 256, 256, 0, stream>>>(W1, w1t);
  }
  {
    int n = KNUM * PNUM * HDIM * HDIM;
    conv_w2<<<(n + 255) / 256, 256, 0, stream>>>(W2, w2t);
  }
  {
    int n = KNUM * PNUM * OUTPAD * HDIM;
    conv_w3<<<(n + 255) / 256, 256, 0, stream>>>(W3, w3t);
  }
  {
    int n = T_TOTAL * PNUM * 9;
    gate_kernel<<<(n + 255) / 256, 256, 0, stream>>>(inp, avoid, out + FP_OFF);
  }
  {
    dim3 grid((T_TOTAL + MTILE - 1) / MTILE, KNUM * PNUM);
    mlp_fused<<<grid, 256, LDS_BYTES, stream>>>(inp, w1t, w2t, w3t, b1, b2, b3, out, predp);
  }
  {
    int n = T_TOTAL * PNUM * 2;
    pred_reduce<<<(n + 255) / 256, 256, 0, stream>>>(predp, out);
  }
}
